// EncoderDecoderEmbeddings_64759516889848
// MI455X (gfx1250) — compile-verified, matrix operand order checked by reading, not run on hardware
//
#include <hip/hip_runtime.h>
#include <cstddef>
#include <cstdint>

typedef __attribute__((ext_vector_type(16))) _Float16 v16h;
typedef __attribute__((ext_vector_type(8)))  _Float16 v8h;
typedef __attribute__((ext_vector_type(4)))  _Float16 v4h;
typedef __attribute__((ext_vector_type(8)))  float    v8f;

#define NTOK 81920      // 512 * 8 * 4 * 5 tokens
#define CDIM 256
#define POSN 160        // 8*4*5 positions (pos_emb broadcast over batch)
#define SIGMA 1.0e-4f
#define LEAK 0.2f

// LDS B-panel row stride in halves: 264 = 132 DWORDs; 132 mod 64 = 4, so the 16
// columns read by a half-wave map to disjoint 4-bank groups (conflict-free b128).
#define LBS 264

// ---------------------------------------------------------------------------
// Weight transpose + f32->f16 convert:  out[b][c][r] = (f16) in[b][r][c]
// (so each WMMA B-column becomes a contiguous row of W^T)
// ---------------------------------------------------------------------------
__global__ void wtrans_kernel(const float* __restrict__ in, _Float16* __restrict__ out,
                              int rows, int cols, int total) {
  int gid = blockIdx.x * blockDim.x + threadIdx.x;
  if (gid >= total) return;
  int per = rows * cols;
  int b   = gid / per;
  int rem = gid - b * per;
  int c   = rem / rows;
  int r   = rem - c * rows;
  out[gid] = (_Float16)in[(size_t)b * per + (size_t)r * cols + c];
}

// ---------------------------------------------------------------------------
// Encoder: h = tanh([x + sigma*noise, rand] @ W_enc) + pos_emb
// ---------------------------------------------------------------------------
__global__ void encode_kernel(const float* __restrict__ x, const float* __restrict__ noise,
                              const float* __restrict__ rnd, const float* __restrict__ Wenc,
                              const float* __restrict__ pos,
                              float* __restrict__ h32, _Float16* __restrict__ h16) {
  int gid = blockIdx.x * blockDim.x + threadIdx.x;   // NTOK * 64
  int n   = gid >> 6;
  int c0  = (gid & 63) << 2;
  float a = x[n] + SIGMA * noise[n];
  float r = rnd[n];
  int  p  = n % POSN;
  float vals[4];
#pragma unroll
  for (int i = 0; i < 4; ++i) {
    int c   = c0 + i;
    vals[i] = tanhf(a * Wenc[c] + r * Wenc[CDIM + c]) + pos[(size_t)p * CDIM + c];
  }
  *(float4*)(h32 + (size_t)n * CDIM + c0) = make_float4(vals[0], vals[1], vals[2], vals[3]);
  v4h hv = {(_Float16)vals[0], (_Float16)vals[1], (_Float16)vals[2], (_Float16)vals[3]};
  *(v4h*)(h16 + (size_t)n * CDIM + c0) = hv;
}

// ---------------------------------------------------------------------------
// WMMA GEMM: C[M x Nout] = A[M x 256] @ W  (W supplied as W^T f16, row = out col)
// Workgroup: 256 threads (8 waves), tile 256(M) x 64(N); wave tile 32 x 64.
// Full 64 x 256 B panel (32KB+pad) staged in LDS ONCE -> single barrier, then a
// fully unrolled sync-free K loop: 8 steps x (2 A frags x 4 B frags) = 64 WMMAs.
// MODE 0: f16 output, no bias.  MODE 1: f32 output + bias, optional accumulate.
// ---------------------------------------------------------------------------
template <int MODE>
__global__ void gemm_kernel(const _Float16* __restrict__ A, const _Float16* __restrict__ WT,
                            _Float16* __restrict__ Oh, float* __restrict__ Of,
                            const float* __restrict__ bias, int Nout, int accum) {
  __shared__ _Float16 lB[64 * LBS];                // padded B panel (~33.8 KB)
  const int lane    = threadIdx.x & 31;
  const int wave    = threadIdx.x >> 5;
  const int mBase   = blockIdx.x * 256 + wave * 32;
  const int nBase   = blockIdx.y * 64;
  const int halfSel = lane >> 4;                   // 0: lanes 0-15, 1: lanes 16-31
  const int l16     = lane & 15;
  const int kOff    = halfSel * 8;                 // ISA 16-bit A/B half-K split

  // ---- stage entire W^T panel (64 rows x 256 halves) into LDS, one barrier ----
#pragma unroll
  for (int it = 0; it < 8; ++it) {
    int idx   = it * 256 + threadIdx.x;            // 2048 chunks of 8 halves
    int row   = idx >> 5;                          // 0..63
    int chunk = (idx & 31) * 8;                    // 0,8,...,248
    *(v8h*)(&lB[row * LBS + chunk]) =
        *(const v8h*)(WT + (size_t)(nBase + row) * CDIM + chunk);
  }
  __syncthreads();

  v8f z = {0.f, 0.f, 0.f, 0.f, 0.f, 0.f, 0.f, 0.f};
  v8f acc0[4] = {z, z, z, z};                      // rows mBase+0..15
  v8f acc1[4] = {z, z, z, z};                      // rows mBase+16..31

  const _Float16* aRow0 = A + (size_t)(mBase + l16) * CDIM + kOff;
  const _Float16* aRow1 = A + (size_t)(mBase + 16 + l16) * CDIM + kOff;

#pragma unroll
  for (int kt = 0; kt < 8; ++kt) {
    const int k0 = kt * 32;
    if (kt < 7) {
      __builtin_prefetch(aRow0 + k0 + 32, 0, 0);
      __builtin_prefetch(aRow1 + k0 + 32, 0, 0);
    }
    // A fragments: lanes 0-15 hold K(0..7,16..23); lanes 16-31 K(8..15,24..31)
    v8h a0lo = *(const v8h*)(aRow0 + k0);
    v8h a0hi = *(const v8h*)(aRow0 + k0 + 16);
    v16h af0 = __builtin_shufflevector(a0lo, a0hi,
        0, 1, 2, 3, 4, 5, 6, 7, 8, 9, 10, 11, 12, 13, 14, 15);
    v8h a1lo = *(const v8h*)(aRow1 + k0);
    v8h a1hi = *(const v8h*)(aRow1 + k0 + 16);
    v16h af1 = __builtin_shufflevector(a1lo, a1hi,
        0, 1, 2, 3, 4, 5, 6, 7, 8, 9, 10, 11, 12, 13, 14, 15);

#pragma unroll
    for (int j = 0; j < 4; ++j) {
      const _Float16* bp = &lB[(j * 16 + l16) * LBS + k0 + kOff];
      v8h blo = *(const v8h*)bp;
      v8h bhi = *(const v8h*)(bp + 16);
      v16h bf = __builtin_shufflevector(blo, bhi,
          0, 1, 2, 3, 4, 5, 6, 7, 8, 9, 10, 11, 12, 13, 14, 15);
      acc0[j] = __builtin_amdgcn_wmma_f32_16x16x32_f16(
          false, af0, false, bf, (short)0, acc0[j], false, false);
      acc1[j] = __builtin_amdgcn_wmma_f32_16x16x32_f16(
          false, af1, false, bf, (short)0, acc1[j], false, false);
    }
  }

  // C/D layout: lane<16 -> rows +0..7, lane>=16 -> rows +8..15; col = l16
#pragma unroll
  for (int half = 0; half < 2; ++half) {
    const int rowB = mBase + half * 16 + halfSel * 8;
    v8f* accp = half ? acc1 : acc0;
#pragma unroll
    for (int j = 0; j < 4; ++j) {
      const int col = nBase + j * 16 + l16;
      if (MODE == 0) {
#pragma unroll
        for (int r = 0; r < 8; ++r)
          Oh[(size_t)(rowB + r) * Nout + col] = (_Float16)accp[j][r];
      } else {
        const float bia = bias[col];
#pragma unroll
        for (int r = 0; r < 8; ++r) {
          size_t idx = (size_t)(rowB + r) * Nout + col;
          float prev = accum ? Of[idx] : 0.0f;
          Of[idx] = prev + accp[j][r] + bia;
        }
      }
    }
  }
}

// ---------------------------------------------------------------------------
// Axial attention core (per token, per head): tiny T (<=8), VALU + exp.
// q: [N][256] f16, kv: [N][512] f16 (k cols 0..255, v cols 256..511)
// ---------------------------------------------------------------------------
__global__ void attn_kernel(const _Float16* __restrict__ q, const _Float16* __restrict__ kv,
                            _Float16* __restrict__ ao, int axis) {
  int gid  = blockIdx.x * blockDim.x + threadIdx.x;  // NTOK * 16
  int head = gid & 15;
  int tmp  = gid >> 4;                               // [0, NTOK)

  int T, stride, base, t;
  if (axis == 0) {
    T = 8; stride = 20;
    t = tmp & 7;
    int seq = tmp >> 3;             // (b, j, k)
    int b = seq / 20, inner = seq % 20;
    base = b * 160 + inner;
  } else if (axis == 1) {
    T = 4; stride = 5;
    t = tmp & 3;
    int seq = tmp >> 2;             // (b, i, k)
    int b = seq / 40, rem = seq % 40;
    base = b * 160 + (rem / 5) * 20 + (rem % 5);
  } else {
    T = 5; stride = 1;
    int seq = tmp / 5;              // (b, i, j)
    t = tmp - seq * 5;
    base = seq * 5;
  }

  const int tokT = base + t * stride;
  float qf[16];
  const _Float16* qp = q + (size_t)tokT * CDIM + head * 16;
#pragma unroll
  for (int d = 0; d < 16; ++d) qf[d] = (float)qp[d];

  float sc[8];
  float mx = -1e30f;
  for (int s = 0; s < T; ++s) {
    const _Float16* kp = kv + (size_t)(base + s * stride) * 512 + head * 16;
    float dot = 0.f;
#pragma unroll
    for (int d = 0; d < 16; ++d) dot += qf[d] * (float)kp[d];
    dot *= 0.25f;                   // DH^-0.5, DH=16
    sc[s] = dot;
    mx = fmaxf(mx, dot);
  }
  float sum = 0.f;
  for (int s = 0; s < T; ++s) { sc[s] = __expf(sc[s] - mx); sum += sc[s]; }
  const float inv = 1.0f / sum;

  float o[16];
#pragma unroll
  for (int d = 0; d < 16; ++d) o[d] = 0.f;
  for (int s = 0; s < T; ++s) {
    const float w = sc[s] * inv;
    const _Float16* vp = kv + (size_t)(base + s * stride) * 512 + 256 + head * 16;
#pragma unroll
    for (int d = 0; d < 16; ++d) o[d] += w * (float)vp[d];
  }
  _Float16* op = ao + (size_t)tokT * CDIM + head * 16;
#pragma unroll
  for (int d = 0; d < 16; ++d) op[d] = (_Float16)o[d];
}

// ---------------------------------------------------------------------------
// Leaky ReLU on the f32 accumulator -> new h32 / h16
// ---------------------------------------------------------------------------
__global__ void lrelu_kernel(const float* __restrict__ acc, float* __restrict__ h32,
                             _Float16* __restrict__ h16) {
  int gid = blockIdx.x * blockDim.x + threadIdx.x;   // NTOK * CDIM / 4
  float4 v = ((const float4*)acc)[gid];
  v.x = v.x >= 0.f ? v.x : LEAK * v.x;
  v.y = v.y >= 0.f ? v.y : LEAK * v.y;
  v.z = v.z >= 0.f ? v.z : LEAK * v.z;
  v.w = v.w >= 0.f ? v.w : LEAK * v.w;
  ((float4*)h32)[gid] = v;
  v4h hv = {(_Float16)v.x, (_Float16)v.y, (_Float16)v.z, (_Float16)v.w};
  ((v4h*)h16)[gid] = hv;
}

// ---------------------------------------------------------------------------
// Decoder: one wave32 per token; out = sigmoid(h . Wd + bd)
// ---------------------------------------------------------------------------
__global__ void decode_kernel(const float* __restrict__ h32, const float* __restrict__ Wd,
                              const float* __restrict__ bd, float* __restrict__ out) {
  int lane = threadIdx.x & 31;
  int wave = threadIdx.x >> 5;
  int n    = blockIdx.x * 8 + wave;
  const float* row = h32 + (size_t)n * CDIM;
  float s = 0.f;
#pragma unroll
  for (int i = 0; i < 8; ++i) {
    int c = lane + i * 32;
    s += row[c] * Wd[c];
  }
#pragma unroll
  for (int off2 = 16; off2 > 0; off2 >>= 1) s += __shfl_xor(s, off2, 32);
  if (lane == 0) out[n] = 1.0f / (1.0f + __expf(-(s + bd[0])));
}

// ---------------------------------------------------------------------------
extern "C" void kernel_launch(void* const* d_in, const int* in_sizes, int n_in,
                              void* d_out, int out_size, void* d_ws, size_t ws_size,
                              hipStream_t stream) {
  (void)in_sizes; (void)n_in; (void)out_size; (void)ws_size;
  const float* x     = (const float*)d_in[0];
  const float* noise = (const float*)d_in[1];
  const float* rnd   = (const float*)d_in[2];
  const float* Wenc  = (const float*)d_in[3];
  const float* pos   = (const float*)d_in[4];
  const float* Wq    = (const float*)d_in[5];   // (2,3,256,256)
  const float* Wkv   = (const float*)d_in[6];   // (2,3,256,512)
  const float* Wo    = (const float*)d_in[7];   // (2,3,256,256)
  const float* bo    = (const float*)d_in[8];   // (2,3,256)
  const float* Wd    = (const float*)d_in[9];   // (256,1)
  const float* bd    = (const float*)d_in[10];  // (1,)
  float* out = (float*)d_out;

  char* ws = (char*)d_ws;
  size_t off = 0;
  auto carve = [&](size_t bytes) -> void* {
    void* p = ws + off;
    off = (off + bytes + 255) & ~(size_t)255;
    return p;
  };
  float*    h32  = (float*)   carve((size_t)NTOK * CDIM * 4);
  float*    accb = (float*)   carve((size_t)NTOK * CDIM * 4);
  _Float16* h16  = (_Float16*)carve((size_t)NTOK * CDIM * 2);
  _Float16* q16  = (_Float16*)carve((size_t)NTOK * CDIM * 2);
  _Float16* kv16 = (_Float16*)carve((size_t)NTOK * CDIM * 4);   // N x 512 f16
  _Float16* ao16 = (_Float16*)carve((size_t)NTOK * CDIM * 2);
  _Float16* wqT  = (_Float16*)carve((size_t)6 * 256 * 256 * 2);
  _Float16* wkvT = (_Float16*)carve((size_t)6 * 512 * 256 * 2);
  _Float16* woT  = (_Float16*)carve((size_t)6 * 256 * 256 * 2);

  // weights -> f16 transposed (W^T rows = output columns, contiguous K)
  {
    int tot = 6 * 256 * 256;
    wtrans_kernel<<<(tot + 255) / 256, 256, 0, stream>>>(Wq, wqT, 256, 256, tot);
    wtrans_kernel<<<(tot + 255) / 256, 256, 0, stream>>>(Wo, woT, 256, 256, tot);
    int tot2 = 6 * 256 * 512;
    wtrans_kernel<<<(tot2 + 255) / 256, 256, 0, stream>>>(Wkv, wkvT, 256, 512, tot2);
  }

  encode_kernel<<<(NTOK * 64) / 256, 256, 0, stream>>>(x, noise, rnd, Wenc, pos, h32, h16);

  const dim3 gQ(NTOK / 256, 256 / 64);   // 320 x 4
  const dim3 gKV(NTOK / 256, 512 / 64);  // 320 x 8

  for (int l = 0; l < 2; ++l) {
    for (int a = 0; a < 3; ++a) {
      const int w = l * 3 + a;
      gemm_kernel<0><<<gQ, 256, 0, stream>>>(
          h16, wqT + (size_t)w * 256 * 256, q16, nullptr, nullptr, 256, 0);
      gemm_kernel<0><<<gKV, 256, 0, stream>>>(
          h16, wkvT + (size_t)w * 512 * 256, kv16, nullptr, nullptr, 512, 0);
      attn_kernel<<<(NTOK * 16) / 256, 256, 0, stream>>>(q16, kv16, ao16, a);
      gemm_kernel<1><<<gQ, 256, 0, stream>>>(
          ao16, woT + (size_t)w * 256 * 256, nullptr, accb,
          bo + (size_t)w * 256, 256, a > 0 ? 1 : 0);
    }
    lrelu_kernel<<<(NTOK * CDIM / 4) / 256, 256, 0, stream>>>(accb, h32, h16);
  }

  decode_kernel<<<NTOK / 8, 256, 0, stream>>>(h32, Wd, bd, out);
}
